// QFormer_2688649527727
// MI455X (gfx1250) — matrix operand order repaired and checked by reading, not compiled, not run on hardware
//
#include <hip/hip_runtime.h>
#include <math.h>

// ---------------- problem dims ----------------
constexpr int BB = 16, NQ = 32, NN = 128, TT = 352, DD = 768, HH = 8;
constexpr int LAYERS = 6, FF = 3072, LL = 512, HD = 96;
constexpr float NEG = -1e9f;
constexpr float SCALE = 0.1020620726159658f;   // 1/sqrt(96)

typedef _Float16 half_t;
typedef __attribute__((ext_vector_type(16))) _Float16 v16h;
typedef __attribute__((ext_vector_type(8)))  _Float16 v8h;
typedef __attribute__((ext_vector_type(8)))  float    v8f;

static __device__ inline v16h cat8(v8h lo, v8h hi) {
    v16h r;
#pragma unroll
    for (int i = 0; i < 8; ++i) { r[i] = lo[i]; r[8 + i] = hi[i]; }
    return r;
}

static __device__ inline v8f wmma_f16(v16h a, v16h b, v8f c) {
    return __builtin_amdgcn_wmma_f32_16x16x32_f16(false, a, false, b, (short)0, c,
                                                  false, false);
}

// async memory -> LDS copy, 16B per lane, tracked by ASYNCcnt
static __device__ inline void async_ld_b128(unsigned lds_off, unsigned voff_bytes,
                                            const void* sbase) {
    asm volatile("global_load_async_to_lds_b128 %0, %1, %2"
                 :: "v"(lds_off), "v"(voff_bytes), "s"(sbase) : "memory");
}

// reduce across the 16-lane group (lanes 0-15 or 16-31)
static __device__ inline float red16_max(float v) {
#pragma unroll
    for (int m = 1; m < 16; m <<= 1) v = fmaxf(v, __shfl_xor(v, m, 32));
    return v;
}
static __device__ inline float red16_sum(float v) {
#pragma unroll
    for (int m = 1; m < 16; m <<= 1) v += __shfl_xor(v, m, 32);
    return v;
}

// ---------------- small utility kernels ----------------

__global__ __launch_bounds__(256) void k_f32_to_f16(const float* __restrict__ in,
                                                    half_t* __restrict__ out, int n) {
    int i = blockIdx.x * 256 + threadIdx.x;
    if (i < n) out[i] = (half_t)in[i];
}

// fp32 W[K][N] -> fp16 Wt[N][K]
__global__ __launch_bounds__(256) void k_convtr(const float* __restrict__ W,
                                                half_t* __restrict__ Wt, int K, int N) {
    int i = blockIdx.x * 256 + threadIdx.x;
    if (i >= K * N) return;
    int k = i / N, n = i - k * N;
    Wt[(size_t)n * K + k] = (half_t)W[i];
}

__global__ __launch_bounds__(256) void k_keep(const int* __restrict__ text_atts,
                                              const int* __restrict__ graph_mask,
                                              int* __restrict__ keep) {
    int i = blockIdx.x * 256 + threadIdx.x;
    if (i >= BB * LL) return;
    int b = i / LL, l = i - b * LL;
    int v;
    if (l < NQ) v = 1;
    else if (l < NQ + NN) v = graph_mask[b * NN + (l - NQ)];
    else v = text_atts[b * TT + (l - NQ - NN)];
    keep[i] = v;
}

__global__ __launch_bounds__(256) void k_embed(const float* __restrict__ qtok,
                                               const float* __restrict__ gfeat,
                                               const float* __restrict__ text,
                                               const float* __restrict__ pos,
                                               const float* __restrict__ tok,
                                               float* __restrict__ pre) {
    int i = blockIdx.x * 256 + threadIdx.x;
    if (i >= BB * LL * DD) return;
    int d = i % DD;
    int bl = i / DD;
    int l = bl % LL, b = bl / LL;
    float s;
    if (l < NQ) s = qtok[l * DD + d];
    else if (l < NQ + NN) s = gfeat[((size_t)b * NN + (l - NQ)) * DD + d];
    else s = text[((size_t)b * TT + (l - NQ - NN)) * DD + d];
    pre[i] = s + pos[l * DD + d] + tok[d];
}

__global__ __launch_bounds__(256) void k_copyout(const float* __restrict__ x,
                                                 float* __restrict__ out) {
    int i = blockIdx.x * 256 + threadIdx.x;
    if (i >= BB * TT * DD) return;
    int d = i % DD;
    int bt = i / DD;
    int t = bt % TT, b = bt / TT;
    out[i] = x[(((size_t)b * LL) + NQ + NN + t) * DD + d];
}

// residual + LayerNorm, emits fp32 x and fp16 xh
__global__ __launch_bounds__(256) void k_ln(const float* __restrict__ y,
                                            const float* __restrict__ res,
                                            const float* __restrict__ g,
                                            const float* __restrict__ bt,
                                            float* __restrict__ xo,
                                            half_t* __restrict__ xh) {
    int row = blockIdx.x;
    size_t base = (size_t)row * DD;
    int tid = threadIdx.x;
    float v[3];
#pragma unroll
    for (int j = 0; j < 3; ++j) {
        int i = tid + j * 256;
        v[j] = y[base + i] + (res ? res[base + i] : 0.f);
    }
    float s = v[0] + v[1] + v[2];
    float s2 = v[0] * v[0] + v[1] * v[1] + v[2] * v[2];
#pragma unroll
    for (int m = 1; m < 32; m <<= 1) { s += __shfl_xor(s, m, 32); s2 += __shfl_xor(s2, m, 32); }
    __shared__ float sh[16];
    int wv = tid >> 5, lane = tid & 31;
    if (lane == 0) { sh[wv] = s; sh[8 + wv] = s2; }
    __syncthreads();
    if (tid == 0) {
        float a = 0.f, b2 = 0.f;
        for (int k = 0; k < 8; ++k) { a += sh[k]; b2 += sh[8 + k]; }
        sh[0] = a; sh[8] = b2;
    }
    __syncthreads();
    float mean = sh[0] * (1.f / DD);
    float var = sh[8] * (1.f / DD) - mean * mean;
    float rs = rsqrtf(var + 1e-12f);
#pragma unroll
    for (int j = 0; j < 3; ++j) {
        int i = tid + j * 256;
        float o = (v[j] - mean) * rs * g[i] + bt[i];
        xo[base + i] = o;
        xh[base + i] = (half_t)o;
    }
}

// ---------------- WMMA GEMM ----------------
// C[M,N] = A[M,K](f16 rowmajor) * Wt[N,K](f16, pre-transposed) + bias
// Block tile 128x256, 8 waves in 2x4 grid, wave tile 64x64 (16 accumulators).
// Tiles staged via GLOBAL_LOAD_ASYNC_TO_LDS_B128 with double buffering.
// mode 0: f32 out; mode 1: GELU->f16; mode 2: QKV scatter (per-head Q,K + V^T)
__global__ __launch_bounds__(256) void k_gemm(const half_t* __restrict__ A,
                                              const half_t* __restrict__ Wt,
                                              const float* __restrict__ bias,
                                              float* __restrict__ outF,
                                              half_t* __restrict__ outH,
                                              half_t* __restrict__ qb,
                                              half_t* __restrict__ kb,
                                              half_t* __restrict__ vtb,
                                              int M, int N, int K, int mode) {
    constexpr int BM = 128, BN = 256, BK = 32, LDT = 40;
    __shared__ half_t As[2][BM * LDT];
    __shared__ half_t Bs[2][BN * LDT];

    const int tid = threadIdx.x;
    const int lane = tid & 31, wv = tid >> 5;
    const int hi = lane >> 4, ln16 = lane & 15;
    const int wm = wv & 1, wn = wv >> 1;          // 2 x 4 wave grid, wave tile 64x64
    const int bM = blockIdx.y, bN = blockIdx.x;

    // per-thread 16B chunks of the tiles
    int aRow[2], aCol[2], bRow[4], bCol[4];
#pragma unroll
    for (int p = 0; p < 2; ++p) {
        int c = (p * 256 + tid) * 8;
        aRow[p] = c >> 5; aCol[p] = c & 31;
    }
#pragma unroll
    for (int p = 0; p < 4; ++p) {
        int c = (p * 256 + tid) * 8;
        bRow[p] = c >> 5; bCol[p] = c & 31;
    }

    auto issue = [&](int buf, int kk) {
#pragma unroll
        for (int p = 0; p < 2; ++p) {
            unsigned lds = (unsigned)(uintptr_t)&As[buf][aRow[p] * LDT + aCol[p]];
            unsigned voff = (unsigned)((((size_t)(bM * BM + aRow[p])) * K + kk + aCol[p]) * 2);
            async_ld_b128(lds, voff, A);
        }
#pragma unroll
        for (int p = 0; p < 4; ++p) {
            unsigned lds = (unsigned)(uintptr_t)&Bs[buf][bRow[p] * LDT + bCol[p]];
            unsigned voff = (unsigned)((((size_t)(bN * BN + bRow[p])) * K + kk + bCol[p]) * 2);
            async_ld_b128(lds, voff, Wt);
        }
    };

    v8f acc[4][4];
    v8f zero = {};
#pragma unroll
    for (int i = 0; i < 4; ++i)
#pragma unroll
        for (int j = 0; j < 4; ++j) acc[i][j] = zero;

    const int nk = K / BK;
    issue(0, 0);
    for (int t = 0; t < nk; ++t) {
        const int buf = t & 1;
        if (t + 1 < nk) {
            issue(buf ^ 1, (t + 1) * BK);
            asm volatile("s_wait_asynccnt 6" ::: "memory");   // batch t landed
        } else {
            asm volatile("s_wait_asynccnt 0" ::: "memory");
        }
        __syncthreads();

        v16h af[4], bf[4];
#pragma unroll
        for (int mt = 0; mt < 4; ++mt) {
            int m = wm * 64 + mt * 16 + ln16;
            af[mt] = cat8(*(const v8h*)(&As[buf][m * LDT + 8 * hi]),
                          *(const v8h*)(&As[buf][m * LDT + 16 + 8 * hi]));
        }
#pragma unroll
        for (int nt = 0; nt < 4; ++nt) {
            int n = wn * 64 + nt * 16 + ln16;
            bf[nt] = cat8(*(const v8h*)(&Bs[buf][n * LDT + 8 * hi]),
                          *(const v8h*)(&Bs[buf][n * LDT + 16 + 8 * hi]));
        }
#pragma unroll
        for (int mt = 0; mt < 4; ++mt)
#pragma unroll
            for (int nt = 0; nt < 4; ++nt)
                acc[mt][nt] = wmma_f16(af[mt], bf[nt], acc[mt][nt]);
        __syncthreads();   // protect buf from next iteration's async writes
    }

    // epilogue
#pragma unroll
    for (int nt = 0; nt < 4; ++nt) {
        int n = bN * BN + wn * 64 + nt * 16 + ln16;
        float bz = bias ? bias[n] : 0.f;
#pragma unroll
        for (int mt = 0; mt < 4; ++mt) {
#pragma unroll
            for (int r = 0; r < 8; ++r) {
                int m = bM * BM + wm * 64 + mt * 16 + r + 8 * hi;
                float v = acc[mt][nt][r] + bz;
                if (mode == 0) {
                    outF[(size_t)m * N + n] = v;
                } else if (mode == 1) {
                    float gelu = 0.5f * v * (1.f + erff(v * 0.70710678118f));
                    outH[(size_t)m * N + n] = (half_t)gelu;
                } else {
                    int which = n / DD;
                    int hd = n - which * DD;
                    int h = hd / HD, d = hd - h * HD;
                    int b = m >> 9, lt = m & (LL - 1);
                    size_t bh = (size_t)b * HH + h;
                    if (which == 0)      qb[(bh * LL + lt) * HD + d] = (half_t)v;
                    else if (which == 1) kb[(bh * LL + lt) * HD + d] = (half_t)v;
                    else                 vtb[(bh * HD + d) * LL + lt] = (half_t)v;
                }
            }
        }
    }
}

// ---------------- flash attention ----------------
// grid (L/128, B*H), 256 threads; wave w owns 16 query rows.
__global__ __launch_bounds__(256) void k_attn(const half_t* __restrict__ Q,
                                              const half_t* __restrict__ Kb,
                                              const half_t* __restrict__ Vt,
                                              const int* __restrict__ keep,
                                              half_t* __restrict__ ctxh) {
    constexpr int LDP = 40;
    __shared__ half_t Ps[8][16 * LDP];

    const int tid = threadIdx.x;
    const int lane = tid & 31, wv = tid >> 5;
    const int hi = lane >> 4, ln16 = lane & 15;
    const int bh = blockIdx.y;
    const int b = bh >> 3, h = bh & 7;
    const int qstart = blockIdx.x * 128 + wv * 16;

    int kq[8];
    float rmax[8], rsum[8];
#pragma unroll
    for (int r = 0; r < 8; ++r) {
        int qi = qstart + r + 8 * hi;
        kq[r] = keep[b * LL + qi];
        rmax[r] = -3.0e38f;
        rsum[r] = 0.f;
    }

    v16h qf[3];
    {
        const half_t* qrow = Q + ((size_t)bh * LL + qstart + ln16) * HD;
#pragma unroll
        for (int c = 0; c < 3; ++c)
            qf[c] = cat8(*(const v8h*)(qrow + c * 32 + 8 * hi),
                         *(const v8h*)(qrow + c * 32 + 16 + 8 * hi));
    }

    v8f cacc[6];
    v8f zero = {};
#pragma unroll
    for (int dt = 0; dt < 6; ++dt) cacc[dt] = zero;

    for (int kt = 0; kt < LL / 32; ++kt) {
        v8f s[2];
#pragma unroll
        for (int sub = 0; sub < 2; ++sub) {
            s[sub] = zero;
            int key = kt * 32 + sub * 16 + ln16;
            const half_t* krow = Kb + ((size_t)bh * LL + key) * HD;
#pragma unroll
            for (int c = 0; c < 3; ++c) {
                v16h kf = cat8(*(const v8h*)(krow + c * 32 + 8 * hi),
                               *(const v8h*)(krow + c * 32 + 16 + 8 * hi));
                s[sub] = wmma_f16(qf[c], kf, s[sub]);
            }
        }
#pragma unroll
        for (int sub = 0; sub < 2; ++sub) {
            int keyn = kt * 32 + sub * 16 + ln16;
            int kk_keep = keep[b * LL + keyn];
            bool is_tk = keyn >= (NQ + NN);
#pragma unroll
            for (int r = 0; r < 8; ++r) {
                int qi = qstart + r + 8 * hi;
                bool is_tq = qi >= (NQ + NN);
                bool causal = (is_tq && is_tk) ? (qi >= keyn) : true;
                bool block_ok = !(!is_tq && is_tk);
                bool allowed = kq[r] && kk_keep && causal && block_ok;
                s[sub][r] = s[sub][r] * SCALE + (allowed ? 0.f : NEG);
            }
        }
        float corr[8];
#pragma unroll
        for (int r = 0; r < 8; ++r) {
            float tmax = red16_max(fmaxf(s[0][r], s[1][r]));
            float nmax = fmaxf(rmax[r], tmax);
            corr[r] = __expf(rmax[r] - nmax);
            float p0 = __expf(s[0][r] - nmax);
            float p1 = __expf(s[1][r] - nmax);
            s[0][r] = p0; s[1][r] = p1;
            float tsum = red16_sum(p0 + p1);
            rsum[r] = rsum[r] * corr[r] + tsum;
            rmax[r] = nmax;
        }
#pragma unroll
        for (int dt = 0; dt < 6; ++dt)
#pragma unroll
            for (int r = 0; r < 8; ++r) cacc[dt][r] *= corr[r];

        // re-layout P (C layout -> A layout) through wave-private LDS
#pragma unroll
        for (int sub = 0; sub < 2; ++sub)
#pragma unroll
            for (int r = 0; r < 8; ++r)
                Ps[wv][(r + 8 * hi) * LDP + sub * 16 + ln16] = (half_t)s[sub][r];
        asm volatile("s_wait_dscnt 0" ::: "memory");
        v16h pf = cat8(*(const v8h*)(&Ps[wv][ln16 * LDP + 8 * hi]),
                       *(const v8h*)(&Ps[wv][ln16 * LDP + 16 + 8 * hi]));

#pragma unroll
        for (int dt = 0; dt < 6; ++dt) {
            int d = dt * 16 + ln16;
            const half_t* vrow = Vt + ((size_t)bh * HD + d) * LL + kt * 32 + 8 * hi;
            v16h vf = cat8(*(const v8h*)(vrow), *(const v8h*)(vrow + 16));
            cacc[dt] = wmma_f16(pf, vf, cacc[dt]);
        }
    }

    float inv[8];
#pragma unroll
    for (int r = 0; r < 8; ++r) inv[r] = 1.f / rsum[r];
#pragma unroll
    for (int dt = 0; dt < 6; ++dt) {
        int d = dt * 16 + ln16;
#pragma unroll
        for (int r = 0; r < 8; ++r) {
            int tokrow = qstart + r + 8 * hi;
            ctxh[((size_t)b * LL + tokrow) * DD + h * HD + d] =
                (half_t)(cacc[dt][r] * inv[r]);
        }
    }
}

// ---------------- host launch ----------------
extern "C" void kernel_launch(void* const* d_in, const int* in_sizes, int n_in,
                              void* d_out, int out_size, void* d_ws, size_t ws_size,
                              hipStream_t stream) {
    (void)in_sizes; (void)n_in; (void)out_size; (void)ws_size;

    const float* graph     = (const float*)d_in[0];
    const float* text      = (const float*)d_in[1];
    const int*   text_atts = (const int*)d_in[2];
    const int*   graph_mask= (const int*)d_in[3];
    const float* qtok      = (const float*)d_in[4];
    const float* gproj_w   = (const float*)d_in[5];
    const float* gproj_b   = (const float*)d_in[6];
    const float* pos_emb   = (const float*)d_in[7];
    const float* tok_emb   = (const float*)d_in[8];
    const float* emb_g     = (const float*)d_in[9];
    const float* emb_b     = (const float*)d_in[10];
    const float* qkv_w     = (const float*)d_in[11];
    const float* qkv_b     = (const float*)d_in[12];
    const float* ao_w      = (const float*)d_in[13];
    const float* ao_b      = (const float*)d_in[14];
    const float* ln1_g     = (const float*)d_in[15];
    const float* ln1_b     = (const float*)d_in[16];
    const float* ff1_w     = (const float*)d_in[17];
    const float* ff1_b     = (const float*)d_in[18];
    const float* ff2_w     = (const float*)d_in[19];
    const float* ff2_b     = (const float*)d_in[20];
    const float* ln2_g     = (const float*)d_in[21];
    const float* ln2_b     = (const float*)d_in[22];
    float* out = (float*)d_out;

    const int M = BB * LL;                       // 8192 tokens
    char* ws = (char*)d_ws;
    size_t off = 0;
    auto alloc = [&](size_t bytes) { size_t r = off; off += (bytes + 255) & ~(size_t)255; return r; };

    float*  x    = (float*)(ws + alloc((size_t)M * DD * 4));
    half_t* xh   = (half_t*)(ws + alloc((size_t)M * DD * 2));
    float*  yf   = (float*)(ws + alloc((size_t)M * DD * 4));
    half_t* hh   = (half_t*)(ws + alloc((size_t)M * FF * 2));   // also reused as f32 pre-embed
    half_t* ctxh = (half_t*)(ws + alloc((size_t)M * DD * 2));   // also reused as f16 graph feats
    half_t* qb   = (half_t*)(ws + alloc((size_t)M * DD * 2));
    half_t* kb   = (half_t*)(ws + alloc((size_t)M * DD * 2));
    half_t* vtb  = (half_t*)(ws + alloc((size_t)M * DD * 2));
    half_t* wt   = (half_t*)(ws + alloc((size_t)DD * FF * 2));
    int*    keep = (int*)(ws + alloc((size_t)BB * LL * 4));

    auto grid1 = [](int n) { return dim3((unsigned)((n + 255) / 256)); };
    const dim3 blk(256);

    k_keep<<<grid1(BB * LL), blk, 0, stream>>>(text_atts, graph_mask, keep);

    // ---- embeddings ----
    k_f32_to_f16<<<grid1(BB * NN * DD), blk, 0, stream>>>(graph, ctxh, BB * NN * DD);
    k_convtr<<<grid1(DD * DD), blk, 0, stream>>>(gproj_w, wt, DD, DD);
    k_gemm<<<dim3(DD / 256, (BB * NN) / 128), blk, 0, stream>>>(
        ctxh, wt, gproj_b, yf, nullptr, nullptr, nullptr, nullptr,
        BB * NN, DD, DD, 0);
    k_embed<<<grid1(M * DD), blk, 0, stream>>>(qtok, yf, text, pos_emb, tok_emb,
                                               (float*)hh);
    k_ln<<<dim3(M), blk, 0, stream>>>((float*)hh, nullptr, emb_g, emb_b, x, xh);

    // ---- transformer layers ----
    for (int i = 0; i < LAYERS; ++i) {
        const float* Wqkv = qkv_w + (size_t)i * DD * 3 * DD;
        const float* bqkv = qkv_b + (size_t)i * 3 * DD;
        const float* Wao  = ao_w  + (size_t)i * DD * DD;
        const float* bao  = ao_b  + (size_t)i * DD;
        const float* W1   = ff1_w + (size_t)i * DD * FF;
        const float* b1   = ff1_b + (size_t)i * FF;
        const float* W2   = ff2_w + (size_t)i * FF * DD;
        const float* b2   = ff2_b + (size_t)i * DD;

        // QKV projection -> per-head Q, K, V^T (f16)
        k_convtr<<<grid1(DD * 3 * DD), blk, 0, stream>>>(Wqkv, wt, DD, 3 * DD);
        k_gemm<<<dim3((3 * DD) / 256, M / 128), blk, 0, stream>>>(
            xh, wt, bqkv, nullptr, nullptr, qb, kb, vtb, M, 3 * DD, DD, 2);

        // flash attention
        k_attn<<<dim3(LL / 128, BB * HH), blk, 0, stream>>>(qb, kb, vtb, keep, ctxh);

        // attention output projection -> yf (f32), then residual+LN
        k_convtr<<<grid1(DD * DD), blk, 0, stream>>>(Wao, wt, DD, DD);
        k_gemm<<<dim3(DD / 256, M / 128), blk, 0, stream>>>(
            ctxh, wt, bao, yf, nullptr, nullptr, nullptr, nullptr, M, DD, DD, 0);
        k_ln<<<dim3(M), blk, 0, stream>>>(yf, x, ln1_g + (size_t)i * DD,
                                          ln1_b + (size_t)i * DD, x, xh);

        // FFN
        k_convtr<<<grid1(DD * FF), blk, 0, stream>>>(W1, wt, DD, FF);
        k_gemm<<<dim3(FF / 256, M / 128), blk, 0, stream>>>(
            xh, wt, b1, nullptr, hh, nullptr, nullptr, nullptr, M, FF, DD, 1);
        k_convtr<<<grid1(FF * DD), blk, 0, stream>>>(W2, wt, FF, DD);
        k_gemm<<<dim3(DD / 256, M / 128), blk, 0, stream>>>(
            hh, wt, b2, yf, nullptr, nullptr, nullptr, nullptr, M, DD, FF, 0);
        k_ln<<<dim3(M), blk, 0, stream>>>(yf, x, ln2_g + (size_t)i * DD,
                                          ln2_b + (size_t)i * DD, x, xh);
    }

    k_copyout<<<grid1(BB * TT * DD), blk, 0, stream>>>(x, out);
}
